// GATEncoder_12721693130769
// MI455X (gfx1250) — matrix operand order repaired and checked by reading, not compiled, not run on hardware
//
#include <hip/hip_runtime.h>
#include <hip/hip_bf16.h>

// ---------------------------------------------------------------------------
// GAT encoder for MI455X (gfx1250, wave32). GEMMs use v_wmma_f32_16x16x32_bf16
// with pre-converted bf16 operands (no per-iteration convert VALU).
// Edge softmax/aggregation are bandwidth-bound scatter ops (float atomics).
// ---------------------------------------------------------------------------

typedef __bf16 bf16x16 __attribute__((ext_vector_type(16)));
typedef float f32x8   __attribute__((ext_vector_type(8)));

union AV { bf16x16 v; uint4 q[2]; };

__device__ __forceinline__ unsigned short f32_to_bf16(float f) {
  unsigned int u = __float_as_uint(f);
  u = (u + 0x7FFFu + ((u >> 16) & 1u)) >> 16;   // RNE round to bf16
  return (unsigned short)u;
}

// ordered-int encoding of float for atomicMax-based segment max
__device__ __forceinline__ int f2ord(float f) {
  int i = __float_as_int(f);
  return i >= 0 ? i : (i ^ 0x7FFFFFFF);
}
__device__ __forceinline__ float ord2f(int i) {
  return __int_as_float(i >= 0 ? i : (i ^ 0x7FFFFFFF));
}

// ---------------------------------------------------------------------------
// WMMA GEMM: C[M,Nc] = A[M,K] * Bt[Nc,K]^T, A/Bt in bf16, C in f32 (or bf16).
// One wave -> one 16x16 tile; inner loop = 4x global_load_b128 + v_wmma.
// flags: bit0 = add bias, bit1 = relu, bit2 = store bf16
// ---------------------------------------------------------------------------
__global__ void wmma_gemm_kernel(const unsigned short* __restrict__ A,
                                 const unsigned short* __restrict__ Bt,
                                 const float* __restrict__ bias,
                                 float* __restrict__ C,
                                 int M, int Nc, int K, int flags) {
  const int wave = threadIdx.x >> 5;
  const int lane = threadIdx.x & 31;
  const int tiles_n = Nc >> 4;
  const int tiles_m = (M + 15) >> 4;
  const int t = blockIdx.x * 8 + wave;
  if (t >= tiles_m * tiles_n) return;          // uniform per wave -> EXEC all 1s
  const int mt = t / tiles_n;
  const int nt = t % tiles_n;
  const int group = lane >> 4;                 // 0 or 1
  const int mrow = mt * 16 + (lane & 15);
  const int mload = (mrow < M) ? mrow : (M - 1);
  const int ncol = nt * 16 + (lane & 15);

  const unsigned short* arow = A  + (size_t)mload * K;
  const unsigned short* brow = Bt + (size_t)ncol  * K;

  f32x8 acc = {0.f, 0.f, 0.f, 0.f, 0.f, 0.f, 0.f, 0.f};

  auto step = [&](int k0) {
    AV a, b;
    // A 16x32 bf16: lane holds M=lane&15; K-chunks [k0+8g .. +7], [k0+16+8g .. +7]
    a.q[0] = *reinterpret_cast<const uint4*>(arow + k0 + group * 8);
    a.q[1] = *reinterpret_cast<const uint4*>(arow + k0 + 16 + group * 8);
    // B 32x16 bf16: lane holds N=lane&15; K-chunks [k0+8g .. +7], [k0+16+8g .. +7]
    b.q[0] = *reinterpret_cast<const uint4*>(brow + k0 + group * 8);
    b.q[1] = *reinterpret_cast<const uint4*>(brow + k0 + 16 + group * 8);
    acc = __builtin_amdgcn_wmma_f32_16x16x32_bf16(
        false, a.v, false, b.v, (short)0, acc, false, false);
  };

  if ((K & 63) == 0) {
    for (int k0 = 0; k0 < K; k0 += 64) { step(k0); step(k0 + 32); }
  } else {
    for (int k0 = 0; k0 < K; k0 += 32) { step(k0); }
  }

  const float bv = (flags & 1) ? bias[ncol] : 0.f;
  #pragma unroll
  for (int r = 0; r < 8; ++r) {
    const int row = mt * 16 + r + 8 * group;   // C/D VGPR layout
    if (row < M) {
      float v = acc[r] + bv;
      if (flags & 2) v = fmaxf(v, 0.f);
      if (flags & 4)
        reinterpret_cast<unsigned short*>(C)[(size_t)row * Nc + ncol] = f32_to_bf16(v);
      else
        C[(size_t)row * Nc + ncol] = v;
    }
  }
}

// Transpose + convert weights: in f32[K][Nc] -> out bf16[Nc][K]
__global__ void transpose_bf16_kernel(const float* __restrict__ in,
                                      unsigned short* __restrict__ out,
                                      int K, int Nc) {
  int t = blockIdx.x * blockDim.x + threadIdx.x;
  if (t >= K * Nc) return;
  int k = t / Nc, n = t % Nc;
  out[(size_t)n * K + k] = f32_to_bf16(in[(size_t)k * Nc + n]);
}

// f32 -> bf16 elementwise
__global__ void cvt_bf16_kernel(const float* __restrict__ in,
                                unsigned short* __restrict__ out, int n) {
  int t = blockIdx.x * blockDim.x + threadIdx.x;
  if (t < n) out[t] = f32_to_bf16(in[t]);
}

// per-node attention logits: als/ald[n,h] = sum_c h[n,h,c]*a_{src,dst}[h,c]
__global__ void attn_logits_kernel(const float* __restrict__ h,
                                   const float* __restrict__ a_src,
                                   const float* __restrict__ a_dst,
                                   float* __restrict__ als,
                                   float* __restrict__ ald, int Nn, int Hh) {
  int t = blockIdx.x * blockDim.x + threadIdx.x;
  if (t >= Nn * Hh) return;
  int hd = t % Hh;
  const float* hp = h + (size_t)t * 64;
  const float* as = a_src + hd * 64;
  const float* ad = a_dst + hd * 64;
  float s = 0.f, d = 0.f;
  for (int c = 0; c < 64; c += 4) {
    float4 hv = *reinterpret_cast<const float4*>(hp + c);
    float4 av = *reinterpret_cast<const float4*>(as + c);
    float4 dv = *reinterpret_cast<const float4*>(ad + c);
    s += hv.x * av.x + hv.y * av.y + hv.z * av.z + hv.w * av.w;
    d += hv.x * dv.x + hv.y * dv.y + hv.z * dv.z + hv.w * dv.w;
  }
  als[t] = s;
  ald[t] = d;
}

__global__ void fill_i32_kernel(int* __restrict__ p, int val, int n) {
  int t = blockIdx.x * blockDim.x + threadIdx.x;
  if (t < n) p[t] = val;
}

// pass 1: e = leaky_relu(als[src]+ald[dst]); segment max over dst (ordered int)
__global__ void edge_logits_max_kernel(const int* __restrict__ ei,
                                       const float* __restrict__ als,
                                       const float* __restrict__ ald,
                                       float* __restrict__ ebuf,
                                       int* __restrict__ mord,
                                       int E, int Nn, int Hh) {
  int t = blockIdx.x * blockDim.x + threadIdx.x;
  int total = E + Nn;
  if (t >= total) return;
  int src, dst;
  if (t < E) { src = ei[t]; dst = ei[E + t]; } else { src = dst = t - E; }
  for (int hd = 0; hd < Hh; ++hd) {
    float v = als[src * Hh + hd] + ald[dst * Hh + hd];
    v = v > 0.f ? v : 0.2f * v;
    ebuf[(size_t)t * Hh + hd] = v;
    atomicMax(&mord[dst * Hh + hd], f2ord(v));
  }
}

// pass 2: p = exp(e - m[dst]); segment sum z
__global__ void edge_exp_sum_kernel(const int* __restrict__ ei,
                                    float* __restrict__ ebuf,
                                    const int* __restrict__ mord,
                                    float* __restrict__ z,
                                    int E, int Nn, int Hh) {
  int t = blockIdx.x * blockDim.x + threadIdx.x;
  int total = E + Nn;
  if (t >= total) return;
  int dst = (t < E) ? ei[E + t] : (t - E);
  for (int hd = 0; hd < Hh; ++hd) {
    float p = __expf(ebuf[(size_t)t * Hh + hd] - ord2f(mord[dst * Hh + hd]));
    ebuf[(size_t)t * Hh + hd] = p;
    atomicAdd(&z[dst * Hh + hd], p);
  }
}

// pass 3: out[dst] += (p / z[dst]) * h[src]   (wave per edge)
__global__ void edge_aggregate_kernel(const int* __restrict__ ei,
                                      const float* __restrict__ pbuf,
                                      const float* __restrict__ z,
                                      const float* __restrict__ hsrc,
                                      float* __restrict__ outm,
                                      int E, int Nn, int Hh) {
  int wave = threadIdx.x >> 5, lane = threadIdx.x & 31;
  int t = blockIdx.x * 8 + wave;
  int total = E + Nn;
  if (t >= total) return;
  int src, dst;
  if (t < E) { src = ei[t]; dst = ei[E + t]; } else { src = dst = t - E; }
  const int HC = Hh * 64;
  for (int idx = lane; idx < HC; idx += 32) {
    int hd = idx >> 6;
    float alpha = pbuf[(size_t)t * Hh + hd] / (z[dst * Hh + hd] + 1e-16f);
    atomicAdd(&outm[(size_t)dst * HC + idx],
              alpha * hsrc[(size_t)src * HC + idx]);
  }
}

// out_bf16 = bf16(relu(in + bias))   (feeds next layer's GEMM directly)
__global__ void bias_act_bf16_kernel(const float* __restrict__ in,
                                     const float* __restrict__ bias,
                                     unsigned short* __restrict__ out,
                                     int n, int cols) {
  int t = blockIdx.x * blockDim.x + threadIdx.x;
  if (t >= n) return;
  out[t] = f32_to_bf16(fmaxf(in[t] + bias[t % cols], 0.f));
}

// out_f32 = in + bias (no relu; layer 3)
__global__ void bias_f32_kernel(const float* __restrict__ in,
                                const float* __restrict__ bias,
                                float* __restrict__ out, int n, int cols) {
  int t = blockIdx.x * blockDim.x + threadIdx.x;
  if (t >= n) return;
  out[t] = in[t] + bias[t % cols];
}

// mean pooling over graphs
__global__ void pool_scatter_kernel(const float* __restrict__ h3,
                                    const int* __restrict__ batch,
                                    float* __restrict__ xg,
                                    float* __restrict__ cnt, int Nn) {
  int t = blockIdx.x * blockDim.x + threadIdx.x;
  if (t >= Nn * 64) return;
  int n = t >> 6, c = t & 63;
  int b = batch[n];
  atomicAdd(&xg[b * 64 + c], h3[t]);
  if (c == 0) atomicAdd(&cnt[b], 1.0f);
}

// xg /= max(cnt,1), emit bf16 for the MLP GEMM
__global__ void pool_div_bf16_kernel(float* __restrict__ xg,
                                     const float* __restrict__ cnt,
                                     unsigned short* __restrict__ xgb, int Bn) {
  int t = blockIdx.x * blockDim.x + threadIdx.x;
  if (t >= Bn * 64) return;
  float v = xg[t] / fmaxf(cnt[t >> 6], 1.0f);
  xg[t] = v;
  xgb[t] = f32_to_bf16(v);
}

// LayerNorm over rows of cols=256, one wave per row
__global__ void layernorm_kernel(const float* __restrict__ y,
                                 const float* __restrict__ g,
                                 const float* __restrict__ b,
                                 float* __restrict__ out, int rows, int cols) {
  int wave = threadIdx.x >> 5, lane = threadIdx.x & 31;
  int row = blockIdx.x * 8 + wave;
  if (row >= rows) return;
  const float* rp = y + (size_t)row * cols;
  float s = 0.f, s2 = 0.f;
  for (int c = lane; c < cols; c += 32) {
    float v = rp[c];
    s += v;
    s2 += v * v;
  }
  for (int off = 16; off; off >>= 1) {
    s  += __shfl_xor(s,  off, 32);
    s2 += __shfl_xor(s2, off, 32);
  }
  float mu  = s / cols;
  float var = s2 / cols - mu * mu;
  float inv = rsqrtf(var + 1e-5f);
  for (int c = lane; c < cols; c += 32)
    out[(size_t)row * cols + c] = (rp[c] - mu) * inv * g[c] + b[c];
}

// ---------------------------------------------------------------------------
// Host side
// ---------------------------------------------------------------------------
static inline int host_f2ord(float f) {
  union { float f; int i; } u; u.f = f;
  return u.i >= 0 ? u.i : (u.i ^ 0x7FFFFFFF);
}

static void launch_gemm(const unsigned short* A, const unsigned short* Bt,
                        const float* bias, float* C, int M, int Nc, int K,
                        int flags, hipStream_t s) {
  int waves = ((M + 15) / 16) * (Nc / 16);
  int blocks = (waves + 7) / 8;
  wmma_gemm_kernel<<<blocks, 256, 0, s>>>(A, Bt, bias, C, M, Nc, K, flags);
}

extern "C" void kernel_launch(void* const* d_in, const int* in_sizes, int n_in,
                              void* d_out, int out_size, void* d_ws, size_t ws_size,
                              hipStream_t stream) {
  constexpr int F = 32, H = 4, C = 64, HC = 256, NHID = 256, NOUT = 256, B = 1000;
  const int N = in_sizes[0] / F;
  const int E = in_sizes[1] / 2;
  const int ET = E + N;  // with self loops

  const float* x     = (const float*)d_in[0];
  const int*   ei    = (const int*)  d_in[1];
  const int*   batch = (const int*)  d_in[2];
  const float* W1 = (const float*)d_in[3];
  const float* as1 = (const float*)d_in[4];
  const float* ad1 = (const float*)d_in[5];
  const float* b1 = (const float*)d_in[6];
  const float* W2 = (const float*)d_in[7];
  const float* as2 = (const float*)d_in[8];
  const float* ad2 = (const float*)d_in[9];
  const float* b2 = (const float*)d_in[10];
  const float* W3 = (const float*)d_in[11];
  const float* as3 = (const float*)d_in[12];
  const float* ad3 = (const float*)d_in[13];
  const float* b3 = (const float*)d_in[14];
  const float* Wm1 = (const float*)d_in[15];
  const float* bm1 = (const float*)d_in[16];
  const float* Wm2 = (const float*)d_in[17];
  const float* bm2 = (const float*)d_in[18];
  const float* g2  = (const float*)d_in[19];
  const float* be2 = (const float*)d_in[20];

  // ---- workspace layout (allocate in 4-byte float units; bf16 use half-slots)
  float* ws = (float*)d_ws;
  size_t off = 0;
  auto alloc_f32  = [&](size_t n) { float* p = ws + off; off += n; return p; };
  auto alloc_bf16 = [&](size_t n) {
    unsigned short* p = (unsigned short*)(ws + off); off += (n + 1) / 2; return p;
  };
  unsigned short* W1t  = alloc_bf16((size_t)F * HC);      // [256][32]
  unsigned short* W2t  = alloc_bf16((size_t)HC * HC);     // [256][256]
  unsigned short* W3t  = alloc_bf16((size_t)HC * C);      // [64][256]
  unsigned short* Wm1t = alloc_bf16((size_t)C * NHID);    // [256][64]
  unsigned short* Wm2t = alloc_bf16((size_t)NHID * NOUT); // [256][256]
  unsigned short* XB   = alloc_bf16((size_t)N * F);       // x in bf16
  unsigned short* HB   = alloc_bf16((size_t)N * HC);      // activations bf16
  unsigned short* XGB  = alloc_bf16((size_t)B * C);
  unsigned short* Y1B  = alloc_bf16((size_t)B * NHID);
  float* ALS  = alloc_f32((size_t)N * H);
  float* ALD  = alloc_f32((size_t)N * H);
  int*   MORD = (int*)alloc_f32((size_t)N * H);
  float* ZB   = alloc_f32((size_t)N * H);
  float* EB   = alloc_f32((size_t)ET * H);
  float* LIN  = alloc_f32((size_t)N * HC);   // GEMM output (pre-bias h)
  float* AGG  = alloc_f32((size_t)N * HC);   // aggregation accumulator
  float* H3   = alloc_f32((size_t)N * C);    // layer-3 output (f32 for pooling)
  float* XG   = alloc_f32((size_t)B * C);
  float* CNT  = alloc_f32((size_t)B);
  float* Y2   = alloc_f32((size_t)B * NOUT);
  (void)ws_size; (void)n_in; (void)out_size;

  const int TPB = 256;
  const int neg_ord = host_f2ord(-1e30f);
  auto grid1 = [&](long n) { return (int)((n + TPB - 1) / TPB); };

  // ---- weight transposes -> bf16 (once per launch; tiny)
  transpose_bf16_kernel<<<grid1((long)F * HC), TPB, 0, stream>>>(W1, W1t, F, HC);
  transpose_bf16_kernel<<<grid1((long)HC * HC), TPB, 0, stream>>>(W2, W2t, HC, HC);
  transpose_bf16_kernel<<<grid1((long)HC * C), TPB, 0, stream>>>(W3, W3t, HC, C);
  transpose_bf16_kernel<<<grid1((long)C * NHID), TPB, 0, stream>>>(Wm1, Wm1t, C, NHID);
  transpose_bf16_kernel<<<grid1((long)NHID * NOUT), TPB, 0, stream>>>(Wm2, Wm2t, NHID, NOUT);
  cvt_bf16_kernel<<<grid1((long)N * F), TPB, 0, stream>>>(x, XB, N * F);

  const int eblocks = grid1(ET);
  const int wblocks = (ET + 7) / 8;  // wave-per-edge kernels

  // ---- GAT layer 1: F -> H*C
  launch_gemm(XB, W1t, nullptr, LIN, N, HC, F, 0, stream);
  attn_logits_kernel<<<grid1((long)N * H), TPB, 0, stream>>>(LIN, as1, ad1, ALS, ALD, N, H);
  fill_i32_kernel<<<grid1((long)N * H), TPB, 0, stream>>>(MORD, neg_ord, N * H);
  hipMemsetAsync(ZB, 0, (size_t)N * H * sizeof(float), stream);
  hipMemsetAsync(AGG, 0, (size_t)N * HC * sizeof(float), stream);
  edge_logits_max_kernel<<<eblocks, TPB, 0, stream>>>(ei, ALS, ALD, EB, MORD, E, N, H);
  edge_exp_sum_kernel<<<eblocks, TPB, 0, stream>>>(ei, EB, MORD, ZB, E, N, H);
  edge_aggregate_kernel<<<wblocks, TPB, 0, stream>>>(ei, EB, ZB, LIN, AGG, E, N, H);
  bias_act_bf16_kernel<<<grid1((long)N * HC), TPB, 0, stream>>>(AGG, b1, HB, N * HC, HC);

  // ---- GAT layer 2: H*C -> H*C
  launch_gemm(HB, W2t, nullptr, LIN, N, HC, HC, 0, stream);
  attn_logits_kernel<<<grid1((long)N * H), TPB, 0, stream>>>(LIN, as2, ad2, ALS, ALD, N, H);
  fill_i32_kernel<<<grid1((long)N * H), TPB, 0, stream>>>(MORD, neg_ord, N * H);
  hipMemsetAsync(ZB, 0, (size_t)N * H * sizeof(float), stream);
  hipMemsetAsync(AGG, 0, (size_t)N * HC * sizeof(float), stream);
  edge_logits_max_kernel<<<eblocks, TPB, 0, stream>>>(ei, ALS, ALD, EB, MORD, E, N, H);
  edge_exp_sum_kernel<<<eblocks, TPB, 0, stream>>>(ei, EB, MORD, ZB, E, N, H);
  edge_aggregate_kernel<<<wblocks, TPB, 0, stream>>>(ei, EB, ZB, LIN, AGG, E, N, H);
  bias_act_bf16_kernel<<<grid1((long)N * HC), TPB, 0, stream>>>(AGG, b2, HB, N * HC, HC);

  // ---- GAT layer 3: H*C -> C, heads=1, no relu
  launch_gemm(HB, W3t, nullptr, LIN, N, C, HC, 0, stream);
  attn_logits_kernel<<<grid1((long)N), TPB, 0, stream>>>(LIN, as3, ad3, ALS, ALD, N, 1);
  fill_i32_kernel<<<grid1((long)N), TPB, 0, stream>>>(MORD, neg_ord, N);
  hipMemsetAsync(ZB, 0, (size_t)N * sizeof(float), stream);
  hipMemsetAsync(AGG, 0, (size_t)N * C * sizeof(float), stream);
  edge_logits_max_kernel<<<eblocks, TPB, 0, stream>>>(ei, ALS, ALD, EB, MORD, E, N, 1);
  edge_exp_sum_kernel<<<eblocks, TPB, 0, stream>>>(ei, EB, MORD, ZB, E, N, 1);
  edge_aggregate_kernel<<<wblocks, TPB, 0, stream>>>(ei, EB, ZB, LIN, AGG, E, N, 1);
  bias_f32_kernel<<<grid1((long)N * C), TPB, 0, stream>>>(AGG, b3, H3, N * C, C);

  // ---- graph mean pooling
  hipMemsetAsync(XG, 0, (size_t)B * C * sizeof(float), stream);
  hipMemsetAsync(CNT, 0, (size_t)B * sizeof(float), stream);
  pool_scatter_kernel<<<grid1((long)N * C), TPB, 0, stream>>>(H3, batch, XG, CNT, N);
  pool_div_bf16_kernel<<<grid1((long)B * C), TPB, 0, stream>>>(XG, CNT, XGB, B);

  // ---- MLP head + LayerNorm
  launch_gemm(XGB, Wm1t, bm1, (float*)Y1B, B, NHID, C, 1 | 2 | 4, stream); // relu -> bf16
  launch_gemm(Y1B, Wm2t, bm2, Y2, B, NOUT, NHID, 1, stream);               // +bias -> f32
  layernorm_kernel<<<(B + 7) / 8, 256, 0, stream>>>(Y2, g2, be2, (float*)d_out, B, NOUT);
}
